// MSARowAttentionWithPairBias_12266426597969
// MI455X (gfx1250) — matrix operand order repaired and checked by reading, not compiled, not run on hardware
//
#include <hip/hip_runtime.h>
#include <hip/hip_bf16.h>

typedef __attribute__((ext_vector_type(16))) _Float16 v16h;
typedef __attribute__((ext_vector_type(8)))  _Float16 v8h;
typedef __attribute__((ext_vector_type(8)))  float    v8f;
typedef __attribute__((ext_vector_type(4)))  float    v4f;

union F16Frag { v16h v; v8h h[2]; };

#define S_DIM   128
#define R_DIM   256
#define CM      256
#define CZ      128
#define NH      8
#define CH      32
#define HC      256      // NH*CH
#define NQKVG   1024     // q|k|v|g concatenated
#define MROWS   32768    // S_DIM * R_DIM
#define VSTR    264      // LDS V^T row stride (f16), multiple of 8 -> 16B aligned rows
#define PSTR    40       // LDS P row stride (f16), multiple of 8

// ---------------- wave reductions (wave32) ----------------
__device__ __forceinline__ float wave_sum32(float v) {
#pragma unroll
  for (int m = 16; m >= 1; m >>= 1) v += __shfl_xor(v, m, 32);
  return v;
}
__device__ __forceinline__ float half_max16(float v) {
#pragma unroll
  for (int m = 8; m >= 1; m >>= 1) v = fmaxf(v, __shfl_xor(v, m, 32));
  return v;
}
__device__ __forceinline__ float half_sum16(float v) {
#pragma unroll
  for (int m = 8; m >= 1; m >>= 1) v += __shfl_xor(v, m, 32);
  return v;
}
__device__ __forceinline__ float fast_rcp(float x) {
  return __builtin_amdgcn_rcpf(x);   // single v_rcp_f32
}

// ---------------- WMMA fragment loaders ----------------
// A 16x32 f16 (row-major source, ld elems/row). Lane L<16: row L, K {0..7,16..23};
// lane L>=16: row L-16, K {8..15,24..31}.
__device__ __forceinline__ v16h load_a16(const _Float16* base, int ld) {
  int lane = threadIdx.x & 31;
  const _Float16* p = base + (size_t)(lane & 15) * ld + ((lane >> 4) << 3);
  F16Frag f;
  f.h[0] = *(const v8h*)p;
  f.h[1] = *(const v8h*)(p + 16);
  return f.v;
}
// B 32x16 f16 from N-major (transposed) source: row = column n, K contiguous.
// Lane L: n = L&15, K {0..7,16..23} (L<16) or {8..15,24..31} (L>=16).
__device__ __forceinline__ v16h load_b16(const _Float16* baseT, int ld) {
  int lane = threadIdx.x & 31;
  const _Float16* p = baseT + (size_t)(lane & 15) * ld + ((lane >> 4) << 3);
  F16Frag f;
  f.h[0] = *(const v8h*)p;
  f.h[1] = *(const v8h*)(p + 16);
  return f.v;
}

// ---------------- K0a: pack w_q|w_k|w_v|w_g -> f16, N-major, fold q scale ----------------
__global__ __launch_bounds__(256) void pack_wqkvg(const float* __restrict__ wq,
                                                  const float* __restrict__ wk,
                                                  const float* __restrict__ wv,
                                                  const float* __restrict__ wg,
                                                  _Float16* __restrict__ wT) {
  int idx = blockIdx.x * 256 + threadIdx.x;  // 0..262143 : n*256 + k
  int n = idx >> 8, k = idx & 255;
  int nl = n & 255;
  const float* src;
  float sc = 1.0f;
  if (n < 256)      { src = wq; sc = 0.17677669529663687f; }  // 1/sqrt(32)
  else if (n < 512) { src = wk; }
  else if (n < 768) { src = wv; }
  else              { src = wg; }
  wT[idx] = (_Float16)(src[k * HC + nl] * sc);
}

// ---------------- K0b: pack w_o -> f16 N-major ----------------
__global__ __launch_bounds__(256) void pack_wo(const float* __restrict__ wo,
                                               _Float16* __restrict__ woT) {
  int idx = blockIdx.x * 256 + threadIdx.x;  // 0..65535 : n*256 + k
  int n = idx >> 8, k = idx & 255;
  woT[idx] = (_Float16)(wo[k * CM + n]);
}

// ---------------- K1: pair bias = LN(z) @ w_z, bias[h][rq][rk] f32 ----------------
__global__ __launch_bounds__(128) void pair_bias(const float* __restrict__ z,
                                                 const float* __restrict__ g,
                                                 const float* __restrict__ b,
                                                 const float* __restrict__ wz,
                                                 float* __restrict__ bias) {
  int lane = threadIdx.x & 31;
  int row  = blockIdx.x * 4 + (threadIdx.x >> 5);   // (rq,rk) flattened, 0..65535
  const float* zr = z + (size_t)row * CZ;
  int c = lane * 4;
  v4f x = *(const v4f*)(zr + c);
  float mu = wave_sum32(x.x + x.y + x.z + x.w) * (1.0f / 128.0f);
  float d0 = x.x - mu, d1 = x.y - mu, d2 = x.z - mu, d3 = x.w - mu;
  float var = wave_sum32(d0 * d0 + d1 * d1 + d2 * d2 + d3 * d3) * (1.0f / 128.0f);
  float rstd = rsqrtf(var + 1e-5f);
  float zn0 = d0 * rstd * g[c + 0] + b[c + 0];
  float zn1 = d1 * rstd * g[c + 1] + b[c + 1];
  float zn2 = d2 * rstd * g[c + 2] + b[c + 2];
  float zn3 = d3 * rstd * g[c + 3] + b[c + 3];
  int rq = row >> 8, rk = row & 255;
#pragma unroll
  for (int h = 0; h < NH; ++h) {
    float p = zn0 * wz[(c + 0) * NH + h] + zn1 * wz[(c + 1) * NH + h] +
              zn2 * wz[(c + 2) * NH + h] + zn3 * wz[(c + 3) * NH + h];
    p = wave_sum32(p);
    if (lane == 0) bias[((size_t)(h << 8) + rq) * 256 + rk] = p;
  }
}

// ---------------- K2: mn = LN(m) -> f16 [32768,256] ----------------
__global__ __launch_bounds__(256) void ln_m_kernel(const float* __restrict__ m,
                                                   const float* __restrict__ g,
                                                   const float* __restrict__ b,
                                                   _Float16* __restrict__ mn) {
  int lane = threadIdx.x & 31;
  int row  = blockIdx.x * 8 + (threadIdx.x >> 5);   // 0..32767
  const float* mr = m + (size_t)row * CM;
  int c = lane * 8;
  v4f x0 = *(const v4f*)(mr + c);
  v4f x1 = *(const v4f*)(mr + c + 4);
  float xs[8] = {x0.x, x0.y, x0.z, x0.w, x1.x, x1.y, x1.z, x1.w};
  float s = 0.f;
#pragma unroll
  for (int j = 0; j < 8; ++j) s += xs[j];
  float mu = wave_sum32(s) * (1.0f / 256.0f);
  float sv = 0.f;
#pragma unroll
  for (int j = 0; j < 8; ++j) { float d = xs[j] - mu; sv += d * d; }
  float rstd = rsqrtf(wave_sum32(sv) * (1.0f / 256.0f) + 1e-5f);
#pragma unroll
  for (int j = 0; j < 8; ++j)
    mn[(size_t)row * CM + c + j] = (_Float16)((xs[j] - mu) * rstd * g[c + j] + b[c + j]);
}

// ---------------- K3: qkvg = mn @ [wq|wk|wv|wg]  (M=32768,N=1024,K=256) ----------------
// grid (128, 16): blockIdx.y = 64-wide N tile (scalar!), 8 waves = 8 M tiles of 32 rows.
// Whole block is gate (n0 >= 768) or not -> scalar branch in the epilogue.
__global__ __launch_bounds__(256) void gemm_qkvg(const _Float16* __restrict__ mn,
                                                 const _Float16* __restrict__ wT,
                                                 const float* __restrict__ bg,
                                                 _Float16* __restrict__ qkvg) {
  int lane = threadIdx.x & 31;
  int m0 = (blockIdx.x * 8 + (threadIdx.x >> 5)) * 32;  // 0..32736
  int n0 = blockIdx.y * 64;                             // scalar, 0..960
  v8f acc[2][4] = {};
  for (int k0 = 0; k0 < 256; k0 += 32) {
    v16h a0 = load_a16(mn + (size_t)m0 * CM + k0, CM);
    v16h a1 = load_a16(mn + (size_t)(m0 + 16) * CM + k0, CM);
    v16h bf[4];
#pragma unroll
    for (int j = 0; j < 4; ++j)
      bf[j] = load_b16(wT + (size_t)(n0 + j * 16) * CM + k0, CM);
#pragma unroll
    for (int j = 0; j < 4; ++j) {
      acc[0][j] = __builtin_amdgcn_wmma_f32_16x16x32_f16(false, a0, false, bf[j],
                                                         (short)0, acc[0][j], false, false);
      acc[1][j] = __builtin_amdgcn_wmma_f32_16x16x32_f16(false, a1, false, bf[j],
                                                         (short)0, acc[1][j], false, false);
    }
  }
  int hb8 = (lane >> 4) << 3, nlo = lane & 15;
  if (n0 >= 768) {   // gate columns: scalar branch, sigmoid epilogue
#pragma unroll
    for (int i = 0; i < 2; ++i)
#pragma unroll
      for (int j = 0; j < 4; ++j) {
        int n = n0 + j * 16 + nlo;
        float bgv = bg[n - 768];   // hoisted: one load per fragment
#pragma unroll
        for (int r = 0; r < 8; ++r) {
          int mrow = m0 + i * 16 + r + hb8;
          float v = fast_rcp(1.0f + __expf(-(acc[i][j][r] + bgv)));
          qkvg[(size_t)mrow * NQKVG + n] = (_Float16)v;
        }
      }
  } else {           // q|k|v columns: plain f16 store
#pragma unroll
    for (int i = 0; i < 2; ++i)
#pragma unroll
      for (int j = 0; j < 4; ++j) {
        int n = n0 + j * 16 + nlo;
#pragma unroll
        for (int r = 0; r < 8; ++r) {
          int mrow = m0 + i * 16 + r + hb8;
          qkvg[(size_t)mrow * NQKVG + n] = (_Float16)acc[i][j][r];
        }
      }
  }
}

// ---------------- K4: attention per (s, h), flash-style online softmax ----------------
// block = 8 waves; wave handles 16 q-rows; blockIdx = (s, h, qhalf)
__global__ __launch_bounds__(256) void attn_kernel(const _Float16* __restrict__ qkvg,
                                                   const float* __restrict__ bias,
                                                   const float* __restrict__ mask,
                                                   _Float16* __restrict__ og) {
  __shared__ __align__(16) _Float16 VT[32 * VSTR];       // V^T [d][rk] for this (s,h)
  __shared__ __align__(16) _Float16 PS[8 * 16 * PSTR];   // per-wave P tile staging
  int s = blockIdx.x, h = blockIdx.y, qh = blockIdx.z;
  int tid = threadIdx.x, lane = tid & 31, w = tid >> 5;
  const _Float16* base = qkvg + (size_t)s * 256 * NQKVG + h * CH;

  // stage V^T into LDS (d = col within head, rk = key row)
  const _Float16* vbase = base + 512;
#pragma unroll 4
  for (int it = 0; it < 32; ++it) {
    int idx = it * 256 + tid;
    int d = idx & 31, rk = idx >> 5;
    VT[d * VSTR + rk] = vbase[(size_t)rk * NQKVG + d];
  }
  __syncthreads();

  int rq0 = qh * 128 + w * 16;
  v16h aq = load_a16(base + (size_t)rq0 * NQKVG, NQKVG);          // Q fragment (pre-scaled)
  const _Float16* kbase = base + 256;
  const float* bb = bias + ((size_t)(h << 8) + rq0) * 256;
  const float* mk = mask + (size_t)s * 256;
  _Float16* myP = PS + w * 16 * PSTR;

  int hb8 = (lane >> 4) << 3, nlo = lane & 15;
  float mrun[8], lrun[8];
#pragma unroll
  for (int r = 0; r < 8; ++r) { mrun[r] = -1e30f; lrun[r] = 0.f; }
  v8f o0 = {}, o1 = {};

  for (int rkc = 0; rkc < 256; rkc += 32) {
    v8f cc[2];
#pragma unroll
    for (int t = 0; t < 2; ++t) {
      int n0 = rkc + t * 16;
      v8f c;
#pragma unroll
      for (int r = 0; r < 8; ++r) c[r] = bb[(size_t)(r + hb8) * 256 + n0 + nlo];  // pair bias
      v16h bk = load_b16(kbase + (size_t)n0 * NQKVG, NQKVG);                      // K^T frag
      c = __builtin_amdgcn_wmma_f32_16x16x32_f16(false, aq, false, bk, (short)0, c, false, false);
      float mb = (mk[n0 + nlo] - 1.0f) * 1e9f;                                    // mask bias
#pragma unroll
      for (int r = 0; r < 8; ++r) c[r] += mb;
      cc[t] = c;
    }
    // online softmax per row
#pragma unroll
    for (int r = 0; r < 8; ++r) {
      float tm = half_max16(fmaxf(cc[0][r], cc[1][r]));
      float mn2 = fmaxf(mrun[r], tm);
      float sc = __expf(mrun[r] - mn2);
      mrun[r] = mn2;
      cc[0][r] = __expf(cc[0][r] - mn2);
      cc[1][r] = __expf(cc[1][r] - mn2);
      float rs = half_sum16(cc[0][r] + cc[1][r]);
      lrun[r] = lrun[r] * sc + rs;
      o0[r] *= sc; o1[r] *= sc;
    }
    // stage P (C-layout -> f16 in LDS) then reload in A-layout
#pragma unroll
    for (int r = 0; r < 8; ++r) {
      int mrow = r + hb8;
      myP[mrow * PSTR + nlo]      = (_Float16)cc[0][r];
      myP[mrow * PSTR + 16 + nlo] = (_Float16)cc[1][r];
    }
    __syncthreads();   // uniform trip count; orders ds writes before cross-lane ds reads
    v16h ap  = load_a16(myP, PSTR);
    v16h bv0 = load_b16(VT + rkc, VSTR);                      // d 0..15
    v16h bv1 = load_b16(VT + (size_t)16 * VSTR + rkc, VSTR);  // d 16..31
    o0 = __builtin_amdgcn_wmma_f32_16x16x32_f16(false, ap, false, bv0, (short)0, o0, false, false);
    o1 = __builtin_amdgcn_wmma_f32_16x16x32_f16(false, ap, false, bv1, (short)0, o1, false, false);
    __syncthreads();
  }
  // normalize, gate with sigmoid projection, store f16
  const _Float16* gbase = base + 768;
#pragma unroll
  for (int r = 0; r < 8; ++r) {
    float inv = fast_rcp(lrun[r]);
    int grow = rq0 + r + hb8;
    float g0 = (float)gbase[(size_t)grow * NQKVG + nlo];
    float g1 = (float)gbase[(size_t)grow * NQKVG + 16 + nlo];
    size_t orow = ((size_t)(s << 8) + grow) * HC + h * CH;
    og[orow + nlo]      = (_Float16)(o0[r] * inv * g0);
    og[orow + 16 + nlo] = (_Float16)(o1[r] * inv * g1);
  }
}

// ---------------- K5: out = og @ w_o + b_o  (M=32768,N=256,K=256), f32 out ----------------
__global__ __launch_bounds__(256) void gemm_out(const _Float16* __restrict__ og,
                                                const _Float16* __restrict__ woT,
                                                const float* __restrict__ bo,
                                                float* __restrict__ out) {
  int lane = threadIdx.x & 31;
  int gw = blockIdx.x * 8 + (threadIdx.x >> 5);   // 0..4095
  int wm = gw >> 2, wn = gw & 3;
  int m0 = wm * 32, n0 = wn * 64;
  v8f acc[2][4] = {};
  for (int k0 = 0; k0 < 256; k0 += 32) {
    v16h a0 = load_a16(og + (size_t)m0 * HC + k0, HC);
    v16h a1 = load_a16(og + (size_t)(m0 + 16) * HC + k0, HC);
    v16h bf[4];
#pragma unroll
    for (int j = 0; j < 4; ++j)
      bf[j] = load_b16(woT + (size_t)(n0 + j * 16) * HC + k0, HC);
#pragma unroll
    for (int j = 0; j < 4; ++j) {
      acc[0][j] = __builtin_amdgcn_wmma_f32_16x16x32_f16(false, a0, false, bf[j],
                                                         (short)0, acc[0][j], false, false);
      acc[1][j] = __builtin_amdgcn_wmma_f32_16x16x32_f16(false, a1, false, bf[j],
                                                         (short)0, acc[1][j], false, false);
    }
  }
  int hb8 = (lane >> 4) << 3, nlo = lane & 15;
#pragma unroll
  for (int i = 0; i < 2; ++i)
#pragma unroll
    for (int j = 0; j < 4; ++j) {
      int n = n0 + j * 16 + nlo;
      float bias_o = bo[n];
#pragma unroll
      for (int r = 0; r < 8; ++r) {
        int mrow = m0 + i * 16 + r + hb8;
        out[(size_t)mrow * CM + n] = acc[i][j][r] + bias_o;
      }
    }
}

// ---------------- launch ----------------
extern "C" void kernel_launch(void* const* d_in, const int* in_sizes, int n_in,
                              void* d_out, int out_size, void* d_ws, size_t ws_size,
                              hipStream_t stream) {
  (void)in_sizes; (void)n_in; (void)out_size; (void)ws_size;
  const float* m      = (const float*)d_in[0];
  const float* z      = (const float*)d_in[1];
  const float* mask   = (const float*)d_in[2];
  const float* ln_m_g = (const float*)d_in[3];
  const float* ln_m_b = (const float*)d_in[4];
  const float* ln_z_g = (const float*)d_in[5];
  const float* ln_z_b = (const float*)d_in[6];
  const float* w_z    = (const float*)d_in[7];
  const float* w_q    = (const float*)d_in[8];
  const float* w_k    = (const float*)d_in[9];
  const float* w_v    = (const float*)d_in[10];
  const float* w_g    = (const float*)d_in[11];
  const float* b_g    = (const float*)d_in[12];
  const float* w_o    = (const float*)d_in[13];
  const float* b_o    = (const float*)d_in[14];
  float* out = (float*)d_out;

  char* p = (char*)d_ws;
  auto carve = [&](size_t bytes) { void* r = (void*)p; p += (bytes + 255) & ~(size_t)255; return r; };
  _Float16* qkvg = (_Float16*)carve((size_t)MROWS * NQKVG * 2);  // 64 MB
  _Float16* mnb  = (_Float16*)carve((size_t)MROWS * CM * 2);     // 16 MB
  _Float16* ogb  = (_Float16*)carve((size_t)MROWS * HC * 2);     // 16 MB
  float*    bias = (float*)carve((size_t)NH * 256 * 256 * 4);    //  2 MB
  _Float16* wT   = (_Float16*)carve((size_t)NQKVG * CM * 2);     // 0.5 MB
  _Float16* woT  = (_Float16*)carve((size_t)CM * HC * 2);        // 0.13 MB

  pack_wqkvg<<<1024, 256, 0, stream>>>(w_q, w_k, w_v, w_g, wT);
  pack_wo<<<256, 256, 0, stream>>>(w_o, woT);
  pair_bias<<<16384, 128, 0, stream>>>(z, ln_z_g, ln_z_b, w_z, bias);
  ln_m_kernel<<<4096, 256, 0, stream>>>(m, ln_m_g, ln_m_b, mnb);
  dim3 qgrid(128, 16);
  gemm_qkvg<<<qgrid, 256, 0, stream>>>(mnb, wT, b_g, qkvg);
  dim3 agrid(S_DIM, NH, 2);
  attn_kernel<<<agrid, 256, 0, stream>>>(qkvg, bias, mask, ogb);
  gemm_out<<<512, 256, 0, stream>>>(ogb, woT, b_o, out);
}
